// RNN_12292196401552
// MI455X (gfx1250) — matrix operand order repaired and checked by reading, not compiled, not run on hardware
//
#include <hip/hip_runtime.h>
#include <hip/hip_bf16.h>
#include <math.h>

typedef _Float16 f16;
typedef __attribute__((ext_vector_type(8)))  _Float16 v8h;
typedef __attribute__((ext_vector_type(16))) _Float16 v16h;
typedef __attribute__((ext_vector_type(8)))  float    v8f;

#define BB    8
#define TT    12
#define HH    128
#define WWID  128
#define HWD   (HH*WWID)
#define HID   32
#define CINX  8
#define TILE  16
#define HALO  18            // TILE + 2 (3x3 SAME halo)
#define NPIX  (HALO*HALO)   // 324
#define KBT   12            // 9 h-taps(32ch) + 3 x-blocks (9 taps x 8ch padded to 96)

// ---------------------------------------------------------------------------
// Weight packing: place W into the WMMA f16 A-fragment layout read by load_afrag
// (K contiguous per lane half: lanes0-15 M=lane, K in {0..15}; lanes16-31 same M rows,
//  K in {16..31}; pairs packed per VGPR).  Semantic K ordering:
//  kb<9  : K = h-channel (0..31) at tap kb           (cin = 8+ch in concat)
//  kb>=9 : q=(kb-9)*32+K -> tap=q/8, x-channel=q%8   (q>=72 padded zero)
// ---------------------------------------------------------------------------
__global__ void pack_weights(const float* __restrict__ Wg,
                             const float* __restrict__ Wc,
                             f16* __restrict__ wgA, f16* __restrict__ wcA)
{
    int idx = blockIdx.x * blockDim.x + threadIdx.x;
    const int NG = 4 * KBT * 32 * 16;   // gates: 4 M-tiles
    const int NC = 2 * KBT * 32 * 16;   // cand : 2 M-tiles
    if (idx >= NG + NC) return;
    const float* src; f16* dst; int li;
    if (idx < NG) { src = Wg; dst = wgA; li = idx; }
    else          { src = Wc; dst = wcA; li = idx - NG; }

    int j    = li & 15;
    int lane = (li >> 4) & 31;
    int frag = li >> 9;                 // mt*KBT + kb
    int kb   = frag % KBT;
    int mt   = frag / KBT;
    int m    = mt * 16 + (lane & 15);
    int half = lane >> 4;
    // A-matrix K contiguous per lane half (matches B fragment): K = 16*half + j
    int kl = half * 16 + j;
    float val = 0.0f;
    if (kb < 9) {
        int cin = 8 + kl;                                // h channels
        val = src[(m * 40 + cin) * 9 + kb];
    } else {
        int q = (kb - 9) * 32 + kl;
        if (q < 72) { int tap = q >> 3, cin = q & 7;     // x channels
                      val = src[(m * 40 + cin) * 9 + tap]; }
    }
    dst[li] = (f16)val;
}

// x (B,8,T,H,W) f32 -> channel-last f16: ((t*B+b)*HW + p)*8 + c  (16B/pixel)
__global__ void pack_x(const float* __restrict__ x, f16* __restrict__ x16)
{
    int idx = blockIdx.x * blockDim.x + threadIdx.x;
    if (idx >= TT * BB * HWD) return;
    int t = idx / (BB * HWD);
    int r = idx - t * (BB * HWD);
    int b = r / HWD;
    int p = r - b * HWD;
    f16 tmp[8] __attribute__((aligned(16)));
#pragma unroll
    for (int c = 0; c < 8; ++c)
        tmp[c] = (f16)x[(((size_t)b * CINX + c) * TT + t) * HWD + p];
    *(int4*)(x16 + (size_t)idx * 8) = *(const int4*)tmp;
}

__global__ void zero_h(f16* __restrict__ h_buf)
{
    int idx = blockIdx.x * blockDim.x + threadIdx.x;
    const int N = BB * HWD * HID / 8;   // int4 granules
    if (idx < N) { int4 z = {0,0,0,0}; *(int4*)(h_buf + (size_t)idx * 8) = z; }
}

// ---------------------------------------------------------------------------
// Fragment loaders
// ---------------------------------------------------------------------------
__device__ __forceinline__ v16h join16(v8h lo, v8h hi) {
    return __builtin_shufflevector(lo, hi, 0,1,2,3,4,5,6,7,8,9,10,11,12,13,14,15);
}

// B-fragment (K x N, lane = pixel column, lane-half selects K range):
// h-taps: 32 contiguous bytes per lane from the 64B/pixel channel-last tile.
// x-blocks: two 16B pixel records at taps t0,t1 (invalid taps -> zeroed slot).
__device__ __forceinline__ v16h load_bfrag(const unsigned char* sh_h,
                                           const unsigned char* sh_x,
                                           int kb, int row, int col, int half)
{
    if (kb < 9) {
        int dy = kb / 3, dx = kb - dy * 3;
        const unsigned char* base =
            sh_h + ((row + dy) * HALO + (col + dx)) * 64 + half * 32;
        return join16(*(const v8h*)base, *(const v8h*)(base + 16));
    } else {
        int t0 = (kb - 9) * 4 + half * 2;
        int t1 = t0 + 1;
        int a0, a1;
        if (t0 < 9) { int dy = t0 / 3, dx = t0 - dy * 3;
                      a0 = ((row + dy) * HALO + (col + dx)) * 16; }
        else a0 = NPIX * 16;                       // zeroed slot
        if (t1 < 9) { int dy = t1 / 3, dx = t1 - dy * 3;
                      a1 = ((row + dy) * HALO + (col + dx)) * 16; }
        else a1 = NPIX * 16;
        return join16(*(const v8h*)(sh_x + a0), *(const v8h*)(sh_x + a1));
    }
}

__device__ __forceinline__ v16h load_afrag(const f16* wA, int mt, int kb, int lane)
{
    const unsigned char* base = (const unsigned char*)wA +
        ((size_t)((mt * KBT + kb) * 32 + lane)) * 32;
    return join16(*(const v8h*)base, *(const v8h*)(base + 16));
}

// Cooperative halo-tile loads (channel-last 64B/pixel and 16B/pixel)
__device__ __forceinline__ void load_tiles(unsigned char* sh_h, unsigned char* sh_x,
                                           const f16* hbuf, const f16* x16,
                                           int b, int t, int tileY, int tileX, int tid)
{
    for (int i = tid; i < NPIX * 4; i += 256) {
        int p = i >> 2, ch = i & 3;
        int r = p / HALO, c = p - r * HALO;
        int gy = tileY * TILE - 1 + r, gx = tileX * TILE - 1 + c;
        int4 v = {0,0,0,0};
        if (gy >= 0 && gy < HH && gx >= 0 && gx < WWID)
            v = *(const int4*)((const unsigned char*)hbuf +
                    ((size_t)(b * HWD + gy * WWID + gx)) * 64 + ch * 16);
        *(int4*)(sh_h + p * 64 + ch * 16) = v;
    }
    for (int i = tid; i < NPIX; i += 256) {
        int r = i / HALO, c = i - r * HALO;
        int gy = tileY * TILE - 1 + r, gx = tileX * TILE - 1 + c;
        int4 v = {0,0,0,0};
        if (gy >= 0 && gy < HH && gx >= 0 && gx < WWID) {
            const unsigned char* src = (const unsigned char*)x16 +
                ((size_t)((t * BB + b) * HWD + gy * WWID + gx)) * 16;
            v = *(const int4*)src;
            if (t + 1 < TT)                       // warm L2 for next timestep
                __builtin_prefetch(src + (size_t)BB * HWD * 16, 0, 1);
        }
        *(int4*)(sh_x + i * 16) = v;
    }
    if (tid == 0) { int4 z = {0,0,0,0}; *(int4*)(sh_x + NPIX * 16) = z; }
    __syncthreads();
}

// ---------------------------------------------------------------------------
// Phase 1: gates = sigmoid(conv([x,h], Wg)+bg); emit rh = r*h and u  (f16)
// ---------------------------------------------------------------------------
__global__ __launch_bounds__(256) void phase1_gates(
    const f16* __restrict__ x16, const f16* __restrict__ wA,
    const float* __restrict__ bias, const f16* __restrict__ h_buf,
    f16* __restrict__ rh_buf, f16* __restrict__ u_buf, int t)
{
    __shared__ __align__(16) unsigned char s_h[NPIX * 64];
    __shared__ __align__(16) unsigned char s_x[NPIX * 16 + 16];
    const int tid = threadIdx.x, blk = blockIdx.x;
    const int b = blk >> 6, tileY = (blk >> 3) & 7, tileX = blk & 7;

    load_tiles(s_h, s_x, h_buf, x16, b, t, tileY, tileX, tid);

    const int wave = tid >> 5, lane = tid & 31;
    const int col = lane & 15, half = lane >> 4;
#pragma unroll
    for (int rr = 0; rr < 2; ++rr) {
        const int row = wave * 2 + rr;
        v8f acc[4] = {v8f{}, v8f{}, v8f{}, v8f{}};
        for (int kb = 0; kb < KBT; ++kb) {
            v16h bf = load_bfrag(s_h, s_x, kb, row, col, half);
#pragma unroll
            for (int mt = 0; mt < 4; ++mt) {
                v16h af = load_afrag(wA, mt, kb, lane);
                acc[mt] = __builtin_amdgcn_wmma_f32_16x16x32_f16(
                    false, af, false, bf, (short)0, acc[mt], false, false);
            }
        }
        const int gy = tileY * TILE + row, gx = tileX * TILE + col;
        const size_t pidx = (size_t)b * HWD + gy * WWID + gx;
        const f16* hpix = (const f16*)(s_h + ((row + 1) * HALO + (col + 1)) * 64);
#pragma unroll
        for (int mt = 0; mt < 4; ++mt) {
#pragma unroll
            for (int v = 0; v < 8; ++v) {
                int cch = mt * 16 + v + 8 * half;          // C-layout: M = v+8*half
                float g = acc[mt][v] + bias[cch];
                float s = 1.0f / (1.0f + __expf(-g));
                if (cch < HID) {                            // reset gate -> r*h
                    rh_buf[pidx * HID + cch] = (f16)(s * (float)hpix[cch]);
                } else {                                    // update gate
                    u_buf[pidx * HID + (cch - HID)] = (f16)s;
                }
            }
        }
    }
}

// ---------------------------------------------------------------------------
// Phase 2: cand = tanh(conv([x,rh], Wc)+bc); h = (1-u)*h + u*cand; emit out
// ---------------------------------------------------------------------------
__global__ __launch_bounds__(256) void phase2_cand(
    const f16* __restrict__ x16, const f16* __restrict__ wA,
    const float* __restrict__ bias, const f16* __restrict__ rh_buf,
    const f16* __restrict__ u_buf, f16* __restrict__ h_buf,
    float* __restrict__ out, int t)
{
    __shared__ __align__(16) unsigned char s_h[NPIX * 64];
    __shared__ __align__(16) unsigned char s_x[NPIX * 16 + 16];
    const int tid = threadIdx.x, blk = blockIdx.x;
    const int b = blk >> 6, tileY = (blk >> 3) & 7, tileX = blk & 7;

    load_tiles(s_h, s_x, rh_buf, x16, b, t, tileY, tileX, tid);

    const int wave = tid >> 5, lane = tid & 31;
    const int col = lane & 15, half = lane >> 4;
#pragma unroll
    for (int rr = 0; rr < 2; ++rr) {
        const int row = wave * 2 + rr;
        v8f acc[2] = {v8f{}, v8f{}};
        for (int kb = 0; kb < KBT; ++kb) {
            v16h bf = load_bfrag(s_h, s_x, kb, row, col, half);
#pragma unroll
            for (int mt = 0; mt < 2; ++mt) {
                v16h af = load_afrag(wA, mt, kb, lane);
                acc[mt] = __builtin_amdgcn_wmma_f32_16x16x32_f16(
                    false, af, false, bf, (short)0, acc[mt], false, false);
            }
        }
        const int gy = tileY * TILE + row, gx = tileX * TILE + col;
        const size_t pidx = (size_t)b * HWD + gy * WWID + gx;
#pragma unroll
        for (int mt = 0; mt < 2; ++mt) {
#pragma unroll
            for (int v = 0; v < 8; ++v) {
                int cch = mt * 16 + v + 8 * half;
                float cand = tanhf(acc[mt][v] + bias[cch]);
                float u  = (float)u_buf[pidx * HID + cch];
                float hp = (float)h_buf[pidx * HID + cch];
                float hn = (1.0f - u) * hp + u * cand;
                h_buf[pidx * HID + cch] = (f16)hn;
                out[(((size_t)b * HID + cch) * TT + t) * HWD + gy * WWID + gx] = hn;
                if (t == TT - 1)                           // h_last (2nd output)
                    out[(size_t)BB * HID * TT * HWD +
                        ((size_t)b * HID + cch) * HWD + gy * WWID + gx] = hn;
            }
        }
    }
}

// ---------------------------------------------------------------------------
extern "C" void kernel_launch(void* const* d_in, const int* in_sizes, int n_in,
                              void* d_out, int out_size, void* d_ws, size_t ws_size,
                              hipStream_t stream)
{
    (void)in_sizes; (void)n_in; (void)out_size; (void)ws_size;
    const float* x  = (const float*)d_in[0];
    const float* Wg = (const float*)d_in[1];
    const float* bg = (const float*)d_in[2];
    const float* Wc = (const float*)d_in[3];
    const float* bc = (const float*)d_in[4];
    float* out = (float*)d_out;

    unsigned char* ws = (unsigned char*)d_ws;
    const size_t SZ_WG = (size_t)4 * KBT * 32 * 16 * sizeof(f16);   // 48 KB
    const size_t SZ_WC = (size_t)2 * KBT * 32 * 16 * sizeof(f16);   // 24 KB
    const size_t SZ_ST = (size_t)BB * HWD * HID * sizeof(f16);      // 8 MB each
    f16* wgA    = (f16*)(ws);
    f16* wcA    = (f16*)(ws + SZ_WG);
    f16* h_buf  = (f16*)(ws + SZ_WG + SZ_WC);
    f16* rh_buf = (f16*)(ws + SZ_WG + SZ_WC + SZ_ST);
    f16* u_buf  = (f16*)(ws + SZ_WG + SZ_WC + 2 * SZ_ST);
    f16* x16    = (f16*)(ws + SZ_WG + SZ_WC + 3 * SZ_ST);

    pack_weights<<<(4*KBT*32*16 + 2*KBT*32*16 + 255) / 256, 256, 0, stream>>>(
        Wg, Wc, wgA, wcA);
    pack_x<<<(TT * BB * HWD + 255) / 256, 256, 0, stream>>>(x, x16);
    zero_h<<<(BB * HWD * HID / 8 + 255) / 256, 256, 0, stream>>>(h_buf);

    for (int t = 0; t < TT; ++t) {
        phase1_gates<<<BB * 8 * 8, 256, 0, stream>>>(
            x16, wgA, bg, h_buf, rh_buf, u_buf, t);
        phase2_cand<<<BB * 8 * 8, 256, 0, stream>>>(
            x16, wcA, bc, rh_buf, u_buf, h_buf, out, t);
    }
}
